// MoE_49967649522233
// MI455X (gfx1250) — matrix operand order repaired and checked by reading, not compiled, not run on hardware
//
#include <hip/hip_runtime.h>
#include <math.h>

typedef __bf16 bf16;
typedef __attribute__((ext_vector_type(8)))  bf16  v8bf;
typedef __attribute__((ext_vector_type(16))) bf16  v16bf;
typedef __attribute__((ext_vector_type(8)))  float v8f;

#define NTOK 4096
#define DIM  1024
#define NEXP 8
#define MAXSLOTS (2*NTOK + NEXP*128)

// ---------------- conversion kernels ----------------

__global__ void f32_to_bf16_kernel(const float* __restrict__ src, bf16* __restrict__ dst, size_t n) {
    size_t i = (size_t)blockIdx.x * blockDim.x + threadIdx.x;
    if (i < n) dst[i] = (bf16)src[i];
}

// dst[b, n, k] = src[b, k, n]  (store W^T in bf16 so WMMA B-operand reads are K-contiguous)
__global__ void transpose_to_bf16_kernel(const float* __restrict__ src, bf16* __restrict__ dst, int batch) {
    size_t i = (size_t)blockIdx.x * blockDim.x + threadIdx.x;
    size_t total = (size_t)batch * DIM * DIM;
    if (i >= total) return;
    int n = (int)(i % DIM);
    int k = (int)((i / DIM) % DIM);
    int b = (int)(i / ((size_t)DIM * DIM));
    dst[(size_t)b * DIM * DIM + (size_t)n * DIM + k] =
        (bf16)src[(size_t)b * DIM * DIM + (size_t)k * DIM + n];
}

// ---------------- WMMA GEMM core ----------------
// C(M,N) = A(M,K) @ B(K,N) + bias(N), B supplied as BT (N x K row-major).
// Block = 256 threads = 8 waves arranged 4(M) x 2(N).
// Wave tile = 32 x 64 (2 M-subtiles x 4 N-subtiles = 8 WMMAs / K-step).
// Block tile = 128 x 128. K-step = 32; ping-pong register double-buffer
// (loop unrolled by 2 so buffer rotation is renaming, not v_mov copies).
// Fragment layouts per CDNA5 ISA 7.12.2 (16-bit A 16x32; B 32x16; 32-bit C 16x16).

__device__ __forceinline__ v16bf load_frag(const bf16* __restrict__ p, int hi_off) {
    v8bf lo = *reinterpret_cast<const v8bf*>(p);
    v8bf hi = *reinterpret_cast<const v8bf*>(p + hi_off);
    return __builtin_shufflevector(lo, hi, 0,1,2,3,4,5,6,7,8,9,10,11,12,13,14,15);
}

template<bool RELU, bool OUT_BF16>
__device__ __forceinline__ void gemm_tile_core(
    const bf16* __restrict__ A,        // row-major, row stride K
    const int*  __restrict__ rowmap,   // nullptr => identity row mapping
    const bf16* __restrict__ BT,       // N x K row-major (i.e. B transposed)
    const float* __restrict__ bias,    // length N
    void* __restrict__ Cout, int ldC,
    int rowBase, int colBase, int K)   // K must be a multiple of 64
{
    const int tid    = threadIdx.x;
    const int wid    = tid >> 5;
    const int lane   = tid & 31;
    const int half   = lane >> 4;
    const int l15    = lane & 15;
    const int m_wave = wid & 3;   // 4 waves along M, 32 rows each  -> block M = 128
    const int n_wave = wid >> 2;  // 2 waves along N, 64 cols each  -> block N = 128

    // A fragment pointers (lane K-offset folded in): lane half h reads K = k+16h..+15 (two 8-elem chunks)
    const bf16* aPtr[2];
#pragma unroll
    for (int mi = 0; mi < 2; ++mi) {
        int outRow = rowBase + m_wave * 32 + mi * 16 + l15;
        int aRow = rowmap ? rowmap[outRow] : outRow;
        aPtr[mi] = A + (size_t)aRow * K + 16 * half;
    }
    // B fragment pointers: lane half h reads K = k+16h..+15 (two 8-elem chunks)
    const bf16* bPtr[4];
#pragma unroll
    for (int t = 0; t < 4; ++t) {
        int col = colBase + n_wave * 64 + t * 16 + l15;
        bPtr[t] = BT + (size_t)col * K + 16 * half;
    }

    v8f acc[2][4];
#pragma unroll
    for (int mi = 0; mi < 2; ++mi)
#pragma unroll
        for (int t = 0; t < 4; ++t)
            acc[mi][t] = (v8f){0.f,0.f,0.f,0.f,0.f,0.f,0.f,0.f};

    auto wmma8 = [&](v16bf (&aa)[2], v16bf (&bb)[4]) {
#pragma unroll
        for (int mi = 0; mi < 2; ++mi)
#pragma unroll
            for (int t = 0; t < 4; ++t)
                acc[mi][t] = __builtin_amdgcn_wmma_f32_16x16x32_bf16(
                    false, aa[mi], false, bb[t], (short)0, acc[mi][t], false, false);
    };

    // ---- ping-pong software pipeline ----
    v16bf a0[2], b0[4], a1[2], b1[4];
#pragma unroll
    for (int mi = 0; mi < 2; ++mi) a0[mi] = load_frag(aPtr[mi], 8);
#pragma unroll
    for (int t = 0; t < 4; ++t)    b0[t]  = load_frag(bPtr[t], 8);

    int k = 32;
    for (; k < K - 32; k += 64) {
        // prefetch step k into buf1, compute with buf0
#pragma unroll
        for (int mi = 0; mi < 2; ++mi) a1[mi] = load_frag(aPtr[mi] + k, 8);
#pragma unroll
        for (int t = 0; t < 4; ++t)    b1[t]  = load_frag(bPtr[t] + k, 8);
        wmma8(a0, b0);
        // prefetch step k+32 into buf0, compute with buf1
#pragma unroll
        for (int mi = 0; mi < 2; ++mi) a0[mi] = load_frag(aPtr[mi] + k + 32, 8);
#pragma unroll
        for (int t = 0; t < 4; ++t)    b0[t]  = load_frag(bPtr[t] + k + 32, 8);
        wmma8(a1, b1);
    }
    // tail: K/32 even => exactly one K-step left to load (k == K-32 here)
#pragma unroll
    for (int mi = 0; mi < 2; ++mi) a1[mi] = load_frag(aPtr[mi] + k, 8);
#pragma unroll
    for (int t = 0; t < 4; ++t)    b1[t]  = load_frag(bPtr[t] + k, 8);
    wmma8(a0, b0);
    wmma8(a1, b1);

    // ---- epilogue: bias (+ReLU), store ----
#pragma unroll
    for (int mi = 0; mi < 2; ++mi) {
#pragma unroll
        for (int t = 0; t < 4; ++t) {
            int col = colBase + n_wave * 64 + t * 16 + l15;
            float bval = bias ? bias[col] : 0.f;
#pragma unroll
            for (int r = 0; r < 8; ++r) {
                int row = rowBase + m_wave * 32 + mi * 16 + half * 8 + r;
                float v = acc[mi][t][r] + bval;
                if (RELU) v = v > 0.f ? v : 0.f;
                if (OUT_BF16) ((bf16*)Cout)[(size_t)row * ldC + col] = (bf16)v;
                else          ((float*)Cout)[(size_t)row * ldC + col] = v;
            }
        }
    }
}

template<bool RELU, bool OUT_BF16>
__global__ void __launch_bounds__(256) gemm_kernel(
    const bf16* __restrict__ A, const bf16* __restrict__ BT,
    const float* __restrict__ bias, void* __restrict__ C, int K, int ldC)
{
    gemm_tile_core<RELU, OUT_BF16>(A, nullptr, BT, bias, C, ldC,
                                   blockIdx.x * 128, blockIdx.y * 128, K);
}

// Expert GEMM: blockIdx.x = e*32 + mtile; rows are routed "slot" rows padded to 128 per expert.
template<bool RELU, bool OUT_BF16, bool INDIRECT>
__global__ void __launch_bounds__(256) expert_gemm_kernel(
    const bf16* __restrict__ A, const int* __restrict__ rowmap,
    const bf16* __restrict__ W, const float* __restrict__ bias,
    void* __restrict__ C, const int* __restrict__ base)
{
    int e  = blockIdx.x >> 5;
    int mt = blockIdx.x & 31;
    int b0 = base[e];
    int b1 = base[e + 1];
    int rowBase = b0 + mt * 128;
    if (rowBase >= b1) return;
    gemm_tile_core<RELU, OUT_BF16>(A, INDIRECT ? rowmap : nullptr,
                                   W + (size_t)e * DIM * DIM, bias + (size_t)e * DIM,
                                   C, DIM, rowBase, blockIdx.y * 128, DIM);
}

// ---------------- gating ----------------
// One wave per token: logits = att@Wg + bg, softmax(8), top-2 (first-index ties), re-softmax of top-2.
__global__ void __launch_bounds__(128) gate_kernel(
    const float* __restrict__ att, const float* __restrict__ Wg, const float* __restrict__ bg,
    float* __restrict__ gate_prob, int* __restrict__ top_e, float* __restrict__ top_w)
{
    int wid  = threadIdx.x >> 5;
    int lane = threadIdx.x & 31;
    int n    = blockIdx.x * 4 + wid;
    if (n >= NTOK) return;

    float s[NEXP];
#pragma unroll
    for (int e = 0; e < NEXP; ++e) s[e] = 0.f;

    const float* ar = att + (size_t)n * DIM;
    for (int kk = lane; kk < DIM; kk += 32) {
        float a = ar[kk];
        const float* wr = Wg + (size_t)kk * NEXP;
#pragma unroll
        for (int e = 0; e < NEXP; ++e) s[e] += a * wr[e];
    }
#pragma unroll
    for (int off = 16; off > 0; off >>= 1) {
#pragma unroll
        for (int e = 0; e < NEXP; ++e) s[e] += __shfl_down(s[e], off, 32);
    }
    if (lane == 0) {
#pragma unroll
        for (int e = 0; e < NEXP; ++e) s[e] += bg[e];
        float mx = s[0];
#pragma unroll
        for (int e = 1; e < NEXP; ++e) mx = s[e] > mx ? s[e] : mx;
        float ex[NEXP], sum = 0.f;
#pragma unroll
        for (int e = 0; e < NEXP; ++e) { ex[e] = __expf(s[e] - mx); sum += ex[e]; }
        float inv = 1.f / sum;
        float p[NEXP];
#pragma unroll
        for (int e = 0; e < NEXP; ++e) { p[e] = ex[e] * inv; gate_prob[(size_t)n * NEXP + e] = p[e]; }
        int i0 = 0;
#pragma unroll
        for (int e = 1; e < NEXP; ++e) if (p[e] > p[i0]) i0 = e;
        int i1 = (i0 == 0) ? 1 : 0;
#pragma unroll
        for (int e = 0; e < NEXP; ++e) if (e != i0 && p[e] > p[i1]) i1 = e;
        float e1 = __expf(p[i1] - p[i0]);       // top-2 re-softmax
        float w0 = 1.f / (1.f + e1);
        float w1 = e1 * w0;
        top_e[2 * n]     = i0;  top_e[2 * n + 1] = i1;
        top_w[2 * n]     = w0;  top_w[2 * n + 1] = w1;
    }
}

// Deterministic per-expert importance sum (fixed-order tree reduction; no atomics).
__global__ void __launch_bounds__(256) importance_kernel(
    const float* __restrict__ gate_prob, float* __restrict__ importance)
{
    __shared__ float sm[256];
    int e = blockIdx.x;
    float s = 0.f;
    for (int n = threadIdx.x; n < NTOK; n += 256) s += gate_prob[(size_t)n * NEXP + e];
    sm[threadIdx.x] = s;
    __syncthreads();
    for (int off = 128; off > 0; off >>= 1) {
        if (threadIdx.x < off) sm[threadIdx.x] += sm[threadIdx.x + off];
        __syncthreads();
    }
    if (threadIdx.x == 0) importance[e] = sm[0];
}

__global__ void loss_kernel(const float* __restrict__ imp, float* __restrict__ out) {
    if (threadIdx.x == 0 && blockIdx.x == 0) {
        float sum = 0.f;
        for (int e = 0; e < NEXP; ++e) sum += imp[e];
        float mean = sum / NEXP;
        float var = 0.f;
        for (int e = 0; e < NEXP; ++e) { float d = imp[e] - mean; var += d * d; }
        var /= (NEXP - 1);   // ddof=1
        out[0] = 0.01f * var / (mean * mean);
    }
}

// ---------------- routing ----------------
__global__ void routing_init_kernel(int* tok_of_row, float* w_of_row, int* counts, int* cursor) {
    int i = blockIdx.x * blockDim.x + threadIdx.x;
    if (i < MAXSLOTS) { tok_of_row[i] = 0; w_of_row[i] = 0.f; }
    if (i < NEXP) { counts[i] = 0; cursor[i] = 0; }
}

__global__ void count_kernel(const int* __restrict__ top_e, int* __restrict__ counts) {
    int n = blockIdx.x * blockDim.x + threadIdx.x;
    if (n >= NTOK) return;
    atomicAdd(&counts[top_e[2 * n]], 1);
    atomicAdd(&counts[top_e[2 * n + 1]], 1);
}

__global__ void scan_kernel(const int* __restrict__ counts, int* __restrict__ base) {
    if (threadIdx.x == 0 && blockIdx.x == 0) {
        int acc = 0;
        base[0] = 0;
        for (int e = 0; e < NEXP; ++e) {
            acc += (counts[e] + 127) & ~127;   // pad each expert to a 128-row block tile
            base[e + 1] = acc;
        }
    }
}

__global__ void assign_kernel(const int* __restrict__ top_e, const float* __restrict__ top_w,
                              const int* __restrict__ base, int* __restrict__ cursor,
                              int* __restrict__ tok_of_row, float* __restrict__ w_of_row,
                              int* __restrict__ slot_map)
{
    int n = blockIdx.x * blockDim.x + threadIdx.x;
    if (n >= NTOK) return;
#pragma unroll
    for (int j = 0; j < 2; ++j) {
        int e = top_e[2 * n + j];
        int pos = atomicAdd(&cursor[e], 1);
        int row = base[e] + pos;
        tok_of_row[row] = n;
        w_of_row[row]   = top_w[2 * n + j];
        slot_map[2 * n + j] = row;
    }
}

// y[n,c] = w0 * out_slots[slot0, c] + w1 * out_slots[slot1, c]
__global__ void combine_kernel(const bf16* __restrict__ out_slots, const int* __restrict__ slot_map,
                               const float* __restrict__ w_of_row, float* __restrict__ y)
{
    size_t i = (size_t)blockIdx.x * blockDim.x + threadIdx.x;
    if (i >= (size_t)NTOK * DIM) return;
    int n = (int)(i / DIM);
    int c = (int)(i % DIM);
    int r0 = slot_map[2 * n], r1 = slot_map[2 * n + 1];
    float v = w_of_row[r0] * (float)out_slots[(size_t)r0 * DIM + c]
            + w_of_row[r1] * (float)out_slots[(size_t)r1 * DIM + c];
    y[i] = v;
}

// ---------------- host launcher ----------------
extern "C" void kernel_launch(void* const* d_in, const int* in_sizes, int n_in,
                              void* d_out, int out_size, void* d_ws, size_t ws_size,
                              hipStream_t stream) {
    const float* x  = (const float*)d_in[0];
    const float* q  = (const float*)d_in[1];
    // d_in[2..5] = Wq,bq,Wk,bk : dead (softmax over singleton axis => attn==1)
    const float* Wv = (const float*)d_in[6];
    const float* bv = (const float*)d_in[7];
    const float* Wo = (const float*)d_in[8];
    const float* bo = (const float*)d_in[9];
    const float* Wg = (const float*)d_in[10];
    const float* bg = (const float*)d_in[11];
    const float* W1 = (const float*)d_in[12];
    const float* b1 = (const float*)d_in[13];
    const float* W2 = (const float*)d_in[14];
    const float* b2 = (const float*)d_in[15];

    char* p = (char*)d_ws;
    auto carve = [&](size_t bytes) -> char* {
        char* r = p;
        p += (bytes + 255) & ~(size_t)255;
        return r;
    };
    bf16*  xb    = (bf16*) carve((size_t)NTOK * DIM * 2);
    bf16*  qb    = (bf16*) carve((size_t)NTOK * DIM * 2);
    bf16*  WvT   = (bf16*) carve((size_t)DIM * DIM * 2);
    bf16*  WoT   = (bf16*) carve((size_t)DIM * DIM * 2);
    bf16*  W1T   = (bf16*) carve((size_t)NEXP * DIM * DIM * 2);
    bf16*  W2T   = (bf16*) carve((size_t)NEXP * DIM * DIM * 2);
    bf16*  Vb    = (bf16*) carve((size_t)NTOK * DIM * 2);
    float* att   = (float*)carve((size_t)NTOK * DIM * 4);
    bf16*  hbuf  = (bf16*) carve((size_t)MAXSLOTS * DIM * 2);
    bf16*  oslot = (bf16*) carve((size_t)MAXSLOTS * DIM * 2);
    int*   tokr  = (int*)  carve((size_t)MAXSLOTS * 4);
    float* wofr  = (float*)carve((size_t)MAXSLOTS * 4);
    int*   smap  = (int*)  carve((size_t)NTOK * 2 * 4);
    int*   tope  = (int*)  carve((size_t)NTOK * 2 * 4);
    float* topw  = (float*)carve((size_t)NTOK * 2 * 4);
    int*   cnts  = (int*)  carve(64);
    int*   curs  = (int*)  carve(64);
    int*   base  = (int*)  carve(64);
    float* imp   = (float*)carve(64);

    float* y_out    = (float*)d_out;                       // (4,1024,1024)
    float* gp_out   = y_out + (size_t)NTOK * DIM;          // (4096, 8)
    float* loss_out = gp_out + (size_t)NTOK * NEXP;        // scalar

    const size_t nd = (size_t)NTOK * DIM;

    // 1) precision conversion + weight transposition
    f32_to_bf16_kernel<<<(int)((nd + 255) / 256), 256, 0, stream>>>(x, xb, nd);
    f32_to_bf16_kernel<<<(int)((nd + 255) / 256), 256, 0, stream>>>(q, qb, nd);
    transpose_to_bf16_kernel<<<(DIM * DIM) / 256, 256, 0, stream>>>(Wv, WvT, 1);
    transpose_to_bf16_kernel<<<(DIM * DIM) / 256, 256, 0, stream>>>(Wo, WoT, 1);
    transpose_to_bf16_kernel<<<(NEXP * DIM * DIM) / 256, 256, 0, stream>>>(W1, W1T, NEXP);
    transpose_to_bf16_kernel<<<(NEXP * DIM * DIM) / 256, 256, 0, stream>>>(W2, W2T, NEXP);

    // 2) V = q@Wv + bv ; att = V@Wo + bo   (attn softmax is identity)
    dim3 ggrid(NTOK / 128, DIM / 128);
    gemm_kernel<false, true ><<<ggrid, 256, 0, stream>>>(qb, WvT, bv, Vb, DIM, DIM);
    gemm_kernel<false, false><<<ggrid, 256, 0, stream>>>(Vb, WoT, bo, att, DIM, DIM);

    // 3) gate probs + top2
    gate_kernel<<<NTOK / 4, 128, 0, stream>>>(att, Wg, bg, gp_out, tope, topw);

    // 4) importance loss (deterministic reduction)
    importance_kernel<<<NEXP, 256, 0, stream>>>(gp_out, imp);
    loss_kernel<<<1, 64, 0, stream>>>(imp, loss_out);

    // 5) routing
    routing_init_kernel<<<(MAXSLOTS + 255) / 256, 256, 0, stream>>>(tokr, wofr, cnts, curs);
    count_kernel<<<NTOK / 256, 256, 0, stream>>>(tope, cnts);
    scan_kernel<<<1, 64, 0, stream>>>(cnts, base);
    assign_kernel<<<NTOK / 256, 256, 0, stream>>>(tope, topw, base, curs, tokr, wofr, smap);

    // 6) routed expert MLP (top-2 only: 4x fewer FLOPs than dense)
    dim3 egrid(NEXP * 32, DIM / 128);
    expert_gemm_kernel<true,  true, true ><<<egrid, 256, 0, stream>>>(xb,   tokr,    W1T, b1, hbuf,  base);
    expert_gemm_kernel<false, true, false><<<egrid, 256, 0, stream>>>(hbuf, nullptr, W2T, b2, oslot, base);

    // 7) weighted combine into y
    combine_kernel<<<(int)(nd / 256), 256, 0, stream>>>(oslot, smap, wofr, y_out);
}
